// MoleculeEncoder_64553358458976
// MI455X (gfx1250) — matrix operand order repaired and checked by reading, not compiled, not run on hardware
//
#include <hip/hip_runtime.h>
#include <hip/hip_bf16.h>

typedef __attribute__((ext_vector_type(16))) __bf16 v16bf;
typedef __attribute__((ext_vector_type(8)))  __bf16 v8bf;
typedef __attribute__((ext_vector_type(4)))  __bf16 v4bf;
typedef __attribute__((ext_vector_type(8)))  float  v8f;

#define NNODES 100000
#define NEDGES 3200000
#define NGRAPH 2048
#define DIM    256
#define EPSV   1e-5f

// ---------------------------------------------------------------- utilities
__global__ void zero_f32_kernel(float* __restrict__ p, int n) {
  int i = blockIdx.x * blockDim.x + threadIdx.x;
  if (i < n) p[i] = 0.0f;
}

__global__ void copy_f32_kernel(const float* __restrict__ src, float* __restrict__ dst) {
  long i = (long)blockIdx.x * blockDim.x + threadIdx.x;
  dst[i] = src[i];
}

__global__ void cvt_f32_bf16_kernel(const float* __restrict__ in, __bf16* __restrict__ out, int n) {
  int i = blockIdx.x * blockDim.x + threadIdx.x;
  if (i < n) out[i] = (__bf16)in[i];
}

// ------------------------------------------------------------- atom encoder
__global__ void atom_encode_kernel(const int* __restrict__ x,
                                   const float* __restrict__ emb,
                                   float* __restrict__ h) {
  int node = blockIdx.x;
  int d    = threadIdx.x;
  float s = 0.0f;
#pragma unroll
  for (int j = 0; j < 9; ++j) {
    int idx = x[node * 9 + j];
    s += emb[((long)(j * 100 + idx)) * DIM + d];
  }
  h[(long)node * DIM + d] = s;
}

// ---------------------------------------------- fused bond-encode + message
#define EPB 8
__global__ void edge_message_kernel(const float* __restrict__ h,
                                    const int* __restrict__ ei,
                                    const int* __restrict__ ea,
                                    const float* __restrict__ bemb,
                                    float* __restrict__ agg) {
  int d = threadIdx.x;
  long e0 = (long)blockIdx.x * EPB;
#pragma unroll
  for (int i = 0; i < EPB; ++i) {
    long e = e0 + i;
    if (e >= NEDGES) return;
    int src = ei[e];
    int dst = ei[NEDGES + e];
    int a0 = ea[e * 3 + 0], a1 = ea[e * 3 + 1], a2 = ea[e * 3 + 2];
    float m = h[(long)src * DIM + d]
            + bemb[(long)(0 * 10 + a0) * DIM + d]
            + bemb[(long)(1 * 10 + a1) * DIM + d]
            + bemb[(long)(2 * 10 + a2) * DIM + d];
    m = fmaxf(m, 0.0f);
    __hip_atomic_fetch_add(&agg[(long)dst * DIM + d], m,
                           __ATOMIC_RELAXED, __HIP_MEMORY_SCOPE_AGENT);
  }
}

// -------------------------------------------------------------- WMMA GEMM
// 256 threads = 8 waves = 4(M) x 2(N) grid of 16x16 WMMA tiles.
// Block tile 64(M) x 32(N), K stepped by 32 through LDS.
// A tile: row-major [64][40] bf16 (80B stride, 16B aligned rows).
// B tile: TRANSPOSED [n=32][k=40] bf16 so B fragments are contiguous-K b128 loads.
// Boundary rows: clamp the source row (garbage rows computed but never stored).
// MODE 0: A f32 [M,K] -> Out bf16 = A@W + bias     (pre-BN activations)
// MODE 1: A bf16 [M,K], fused y = relu(y*scale[k]+shift[k]) during staging,
//         Out f32 = relu(A'@W + bias)
#define BM 64
#define BN 32
#define BK 32
#define ASTR 40
#define BTSTR 40

template <int MODE>
__global__ __launch_bounds__(256)
void gemm_wmma_kernel(const void* __restrict__ Ain,
                      const __bf16* __restrict__ W,
                      const float* __restrict__ bias,
                      const float* __restrict__ scale,
                      const float* __restrict__ shift,
                      void* __restrict__ Out,
                      int M, int K, int N) {
  __shared__ __attribute__((aligned(16))) __bf16 Alds[BM * ASTR];
  __shared__ __attribute__((aligned(16))) __bf16 Btld[BN * BTSTR];

  const int tid  = threadIdx.x;
  const int lane = tid & 31;
  const int warp = tid >> 5;
  const int wr = warp & 3;
  const int wc = warp >> 2;
  const int m0 = blockIdx.y * BM;
  const int n0 = blockIdx.x * BN;

  // staging coordinates (computed once)
  const int arow = tid >> 2;           // 0..63
  const int acg  = (tid & 3) * 8;      // 0,8,16,24
  long agm = (long)m0 + arow;
  if (agm > M - 1) agm = M - 1;        // clamp: no predication, rows>=M never stored
  const int bkk = tid >> 3;            // 0..31
  const int bcg = (tid & 7) * 4;       // 0,4,...,28

  // fragment coordinates
  const int am    = wr * 16 + (lane & 15);
  const int khalf = (lane >= 16) ? 8 : 0;
  const int bn    = wc * 16 + (lane & 15);
  const int koff  = (lane >= 16) ? 8 : 0;

  v8f acc = {};

  for (int k0 = 0; k0 < K; k0 += BK) {
    // ---- stage A tile: one 16B LDS store per thread
    if (MODE == 0) {
      const float* A = (const float*)Ain;
      const float4* src = (const float4*)&A[agm * K + k0 + acg];
      float4 f0 = src[0];
      float4 f1 = src[1];
      v8bf pk;
      pk[0] = (__bf16)f0.x; pk[1] = (__bf16)f0.y; pk[2] = (__bf16)f0.z; pk[3] = (__bf16)f0.w;
      pk[4] = (__bf16)f1.x; pk[5] = (__bf16)f1.y; pk[6] = (__bf16)f1.z; pk[7] = (__bf16)f1.w;
      *(v8bf*)&Alds[arow * ASTR + acg] = pk;
    } else {
      const __bf16* A = (const __bf16*)Ain;
      v8bf raw = *(const v8bf*)&A[agm * K + k0 + acg];
      v8bf pk;
#pragma unroll
      for (int u = 0; u < 8; ++u) {
        int kk = k0 + acg + u;
        float v = fmaxf((float)raw[u] * scale[kk] + shift[kk], 0.0f);  // fused BN+ReLU
        pk[u] = (__bf16)v;
      }
      *(v8bf*)&Alds[arow * ASTR + acg] = pk;
    }
    // ---- stage B tile transposed: 8B global load, 4 scalar LDS stores
    {
      v4bf w4 = *(const v4bf*)&W[(long)(k0 + bkk) * N + n0 + bcg];
#pragma unroll
      for (int u = 0; u < 4; ++u)
        Btld[(bcg + u) * BTSTR + bkk] = w4[u];
    }
    __syncthreads();

    // ---- fragments: contiguous-K b128 loads per documented CDNA5 layout
    v8bf a_lo = *(const v8bf*)&Alds[am * ASTR + khalf];        // e=0..7  : K=khalf..+7
    v8bf a_hi = *(const v8bf*)&Alds[am * ASTR + khalf + 16];   // e=8..15 : K=khalf+16..+23
    v8bf b_lo = *(const v8bf*)&Btld[bn * BTSTR + koff];        // e=0..7  : K=koff..+7
    v8bf b_hi = *(const v8bf*)&Btld[bn * BTSTR + koff + 16];   // e=8..15 : K=koff+16..+23
    v16bf a = __builtin_shufflevector(a_lo, a_hi, 0,1,2,3,4,5,6,7,8,9,10,11,12,13,14,15);
    v16bf b = __builtin_shufflevector(b_lo, b_hi, 0,1,2,3,4,5,6,7,8,9,10,11,12,13,14,15);

    acc = __builtin_amdgcn_wmma_f32_16x16x32_bf16(
        false, a, false, b, (short)0, acc, false, false);

    __syncthreads();
  }

  // ---- epilogue: VGPR r -> M = r + 8*(lane>=16), N = lane&15
  int n     = n0 + wc * 16 + (lane & 15);
  int mbase = m0 + wr * 16 + ((lane >= 16) ? 8 : 0);
  float bv = bias[n];
#pragma unroll
  for (int r = 0; r < 8; ++r) {
    int m = mbase + r;
    if (m < M) {
      float v = acc[r] + bv;
      if (MODE == 0) {
        ((__bf16*)Out)[(long)m * N + n] = (__bf16)v;
      } else {
        ((float*)Out)[(long)m * N + n] = fmaxf(v, 0.0f);
      }
    }
  }
}

// ------------------------------------------------------ BatchNorm statistics
__global__ void bn_stats_kernel(const __bf16* __restrict__ Y,
                                float* __restrict__ sum,
                                float* __restrict__ sumsq,
                                int M, int N, int rowsPerBlock) {
  int c = blockIdx.x * blockDim.x + threadIdx.x;
  if (c >= N) return;
  long r0 = (long)blockIdx.y * rowsPerBlock;
  long r1 = r0 + rowsPerBlock;
  if (r1 > M) r1 = M;
  float s = 0.0f, s2 = 0.0f;
  for (long r = r0; r < r1; ++r) {
    float v = (float)Y[r * N + c];
    s += v;
    s2 += v * v;
  }
  __hip_atomic_fetch_add(&sum[c],   s,  __ATOMIC_RELAXED, __HIP_MEMORY_SCOPE_AGENT);
  __hip_atomic_fetch_add(&sumsq[c], s2, __ATOMIC_RELAXED, __HIP_MEMORY_SCOPE_AGENT);
}

__global__ void bn_finalize_kernel(const float* __restrict__ sum,
                                   const float* __restrict__ sumsq,
                                   const float* __restrict__ g,
                                   const float* __restrict__ be,
                                   float* __restrict__ scale,
                                   float* __restrict__ shift,
                                   int N, float invM) {
  int c = blockIdx.x * blockDim.x + threadIdx.x;
  if (c >= N) return;
  float mu  = sum[c] * invM;
  float var = sumsq[c] * invM - mu * mu;
  float sc  = g[c] * rsqrtf(var + EPSV);
  scale[c] = sc;
  shift[c] = be[c] - mu * sc;
}

// ----------------------------------------------------------- graph pooling
__global__ void pool_kernel(const float* __restrict__ h,
                            const int* __restrict__ batch,
                            float* __restrict__ out) {
  int node = blockIdx.x;
  int d    = threadIdx.x;
  int g    = batch[node];
  __hip_atomic_fetch_add(&out[(long)g * DIM + d], h[(long)node * DIM + d],
                         __ATOMIC_RELAXED, __HIP_MEMORY_SCOPE_AGENT);
}

// ------------------------------------------------------------------- driver
extern "C" void kernel_launch(void* const* d_in, const int* in_sizes, int n_in,
                              void* d_out, int out_size, void* d_ws, size_t ws_size,
                              hipStream_t stream) {
  (void)in_sizes; (void)n_in; (void)out_size; (void)ws_size;

  const int*   x      = (const int*)  d_in[0];
  const int*   ei     = (const int*)  d_in[1];
  const int*   ea     = (const int*)  d_in[2];
  const int*   batch  = (const int*)  d_in[3];
  const float* aemb   = (const float*)d_in[4];
  const float* bemb1  = (const float*)d_in[5];
  const float* bemb2  = (const float*)d_in[6];
  const float* c1_w1  = (const float*)d_in[7];
  const float* c1_b1  = (const float*)d_in[8];
  const float* c1_g   = (const float*)d_in[9];
  const float* c1_be  = (const float*)d_in[10];
  const float* c1_w2  = (const float*)d_in[11];
  const float* c1_b2  = (const float*)d_in[12];
  const float* c2_w1  = (const float*)d_in[13];
  const float* c2_b1  = (const float*)d_in[14];
  const float* c2_g   = (const float*)d_in[15];
  const float* c2_be  = (const float*)d_in[16];
  const float* c2_w2  = (const float*)d_in[17];
  const float* c2_b2  = (const float*)d_in[18];
  float* out = (float*)d_out;

  // ---- workspace carve-up
  float*  h    = (float*)d_ws;                              // [N, 256] f32
  float*  agg  = h   + (size_t)NNODES * DIM;                // [N, 256] f32
  __bf16* Y    = (__bf16*)(agg + (size_t)NNODES * DIM);     // [N, 512] bf16
  __bf16* w1a  = Y   + (size_t)NNODES * 2 * DIM;            // [256,512] bf16
  __bf16* w1b  = w1a + (size_t)DIM * 2 * DIM;               // [512,256] bf16
  __bf16* w2a  = w1b + (size_t)2 * DIM * DIM;               // [256,256] bf16
  __bf16* w2b  = w2a + (size_t)DIM * DIM;                   // [256,256] bf16
  float*  stats = (float*)(w2b + (size_t)DIM * DIM);
  float*  ssum  = stats;
  float*  ssq   = stats + 512;
  float*  scl   = stats + 1024;
  float*  shf   = stats + 1536;

  const int M = NNODES;
  const int rpb = (M + 255) / 256;
  const float invM = 1.0f / (float)M;
  const long nodeElems = (long)NNODES * DIM;

  zero_f32_kernel<<<(NGRAPH * DIM + 255) / 256, 256, 0, stream>>>(out, NGRAPH * DIM);

  cvt_f32_bf16_kernel<<<(DIM * 2 * DIM + 255) / 256, 256, 0, stream>>>(c1_w1, w1a, DIM * 2 * DIM);
  cvt_f32_bf16_kernel<<<(2 * DIM * DIM + 255) / 256, 256, 0, stream>>>(c1_w2, w1b, 2 * DIM * DIM);
  cvt_f32_bf16_kernel<<<(DIM * DIM + 255) / 256, 256, 0, stream>>>(c2_w1, w2a, DIM * DIM);
  cvt_f32_bf16_kernel<<<(DIM * DIM + 255) / 256, 256, 0, stream>>>(c2_w2, w2b, DIM * DIM);

  atom_encode_kernel<<<NNODES, DIM, 0, stream>>>(x, aemb, h);

  const dim3 gemmBlk(256);
  const int edgeGrid = (NEDGES + EPB - 1) / EPB;

  // =========================== conv 1 ===========================
  copy_f32_kernel<<<(unsigned)(nodeElems / 256), 256, 0, stream>>>(h, agg);
  edge_message_kernel<<<edgeGrid, DIM, 0, stream>>>(h, ei, ea, bemb1, agg);

  zero_f32_kernel<<<(1024 + 255) / 256, 256, 0, stream>>>(stats, 1024);
  gemm_wmma_kernel<0><<<dim3((2 * DIM) / BN, (M + BM - 1) / BM), gemmBlk, 0, stream>>>(
      agg, w1a, c1_b1, nullptr, nullptr, Y, M, DIM, 2 * DIM);
  bn_stats_kernel<<<dim3((2 * DIM) / 256, 256), 256, 0, stream>>>(Y, ssum, ssq, M, 2 * DIM, rpb);
  bn_finalize_kernel<<<2, 256, 0, stream>>>(ssum, ssq, c1_g, c1_be, scl, shf, 2 * DIM, invM);
  gemm_wmma_kernel<1><<<dim3(DIM / BN, (M + BM - 1) / BM), gemmBlk, 0, stream>>>(
      Y, w1b, c1_b2, scl, shf, h, M, 2 * DIM, DIM);

  // =========================== conv 2 ===========================
  copy_f32_kernel<<<(unsigned)(nodeElems / 256), 256, 0, stream>>>(h, agg);
  edge_message_kernel<<<edgeGrid, DIM, 0, stream>>>(h, ei, ea, bemb2, agg);

  zero_f32_kernel<<<(1024 + 255) / 256, 256, 0, stream>>>(stats, 1024);
  gemm_wmma_kernel<0><<<dim3(DIM / BN, (M + BM - 1) / BM), gemmBlk, 0, stream>>>(
      agg, w2a, c2_b1, nullptr, nullptr, Y, M, DIM, DIM);
  bn_stats_kernel<<<dim3(DIM / 256, 256), 256, 0, stream>>>(Y, ssum, ssq, M, DIM, rpb);
  bn_finalize_kernel<<<1, 256, 0, stream>>>(ssum, ssq, c2_g, c2_be, scl, shf, DIM, invM);
  gemm_wmma_kernel<1><<<dim3(DIM / BN, (M + BM - 1) / BM), gemmBlk, 0, stream>>>(
      Y, w2b, c2_b2, scl, shf, h, M, DIM, DIM);

  pool_kernel<<<NNODES, DIM, 0, stream>>>(h, batch, out);
}
